// PitchEnergyPredictor_27161373179990
// MI455X (gfx1250) — compile-verified, matrix operand order checked by reading, not run on hardware
//
#include <hip/hip_runtime.h>
#include <math.h>

// Problem constants (from reference)
#define BB 8
#define TT 256
#define FF 1024
#define CC 576
#define SS 64
#define HH 8
#define DD 72   // head dim = CC/HH
#define WIN 5

typedef __attribute__((ext_vector_type(16))) _Float16 v16h;
typedef __attribute__((ext_vector_type(4)))  _Float16 v4h;
typedef __attribute__((ext_vector_type(8)))  float    v8f;
typedef __attribute__((ext_vector_type(4)))  float    v4f;

// ---------------------------------------------------------------------------
// LDS-staged batched GEMM:
//   C[z,m,n] = alpha * (bias[m] + Res[z,m,n] +
//              sum_r sum_k A[z,r,m,k] * B[z,k,n + r - pad])
// A element addr: A + z*a_bs + r*a_rs + m*a_is + k*a_ks
// B element addr: B + z*b_bs + (n+r-pad)*b_is + k*b_ks
//
// Block: 128 threads (4 waves) -> one 16(M) x 64(N) strip.
//   - A k-slab (16x32) staged once per block into LDS, fragment-ordered.
//   - Each wave owns one 16-col B tile (32x16), also fragment-ordered.
//   - Thread t stages fragment elements [4t, 4t+4): same lane-row, 4
//     consecutive K -> one ds_store_b64; and for unit-K-stride operands
//     (weights, attn-V) one global_load_b128.
//   - OOB handled by index clamping + value zeroing (no exec divergence;
//     EXEC is all-ones at every v_wmma as the ISA requires).
//   - Fragment read = contiguous 32B per lane -> 2x ds_load_b128.
// Fragment order (16-bit operands, wave32):
//   lane l: row/col = l&15, kb = 8*(l>>4); elem j: k = kb + j (j<8), kb+8+j (j>=8)
// ---------------------------------------------------------------------------
__global__ void wmma_gemm_kernel(const float* __restrict__ A, const float* __restrict__ Bm,
                                 const float* __restrict__ bias, const float* __restrict__ Res,
                                 float* __restrict__ Cm,
                                 int M, int N, int K,
                                 long a_is, long a_ks, long a_bs, long a_rs,
                                 long b_is, long b_ks, long b_bs,
                                 long c_bs, int R, int pad, float alpha) {
  __shared__ alignas(32) _Float16 As[512];        // 16x32 A slab, fragment order
  __shared__ alignas(32) _Float16 Bs[4 * 512];    // 4 tiles of 32x16 B, fragment order

  int tid  = threadIdx.x;
  int wave = tid >> 5;
  int lane = tid & 31;
  int li   = lane & 31 & 15;
  int m0   = blockIdx.y * 16;
  int nblk = blockIdx.x * 64;                     // N % 64 == 0 for every call site
  int z    = blockIdx.z;
  const float* Az = A  + (long)z * a_bs;
  const float* Bz = Bm + (long)z * b_bs;

  // per-thread staging geometry: 4 consecutive fragment elements
  int e0 = tid << 2;                              // 0..508
  int sl = e0 >> 4;                               // staging lane 0..31
  int j0 = e0 & 15;                               // 0,4,8,12
  int srow = sl & 15;                             // A row / B col within tile
  int kfb  = ((sl & 16) >> 1) + j0 + ((j0 & 8) ? 8 : 0);  // k offset of elem 0

  v8f acc = {};
  for (int r = 0; r < R; ++r) {
    const float* Ar = Az + (long)r * a_rs;
    for (int k0 = 0; k0 < K; k0 += 32) {
      if (k0 + 32 < K) {                          // prefetch next k-slab
        __builtin_prefetch(Ar + (long)m0 * a_is + (long)(k0 + 32) * a_ks, 0, 1);
        __builtin_prefetch(Bz + (long)(nblk + r - pad) * b_is + (long)(k0 + 32) * b_ks, 0, 1);
      }
      int kk0 = k0 + kfb;
      // ---- stage A: 4 elems/thread, one ds_store_b64 ----
      {
        int m = m0 + srow;
        int mc = (m < M) ? m : (M - 1);
        const float* ap = Ar + (long)mc * a_is;
        bool mv = (m < M);
        v4h hv;
        if (a_ks == 1) {                          // contiguous K: global_load_b128
          int kc = (kk0 <= K - 4) ? kk0 : ((K >= 4) ? K - 4 : 0);
          v4f f4 = *(const v4f*)(ap + kc);
#pragma unroll
          for (int i = 0; i < 4; ++i) {
            bool v = mv && (kk0 + i < K);
            hv[i] = (_Float16)(v ? f4[i] : 0.0f);
          }
        } else {
          float f[4];
#pragma unroll
          for (int i = 0; i < 4; ++i) {
            int kk = kk0 + i;
            int kc = (kk < K) ? kk : (K - 1);
            f[i] = ap[(long)kc * a_ks];
          }
#pragma unroll
          for (int i = 0; i < 4; ++i) {
            bool v = mv && (kk0 + i < K);
            hv[i] = (_Float16)(v ? f[i] : 0.0f);
          }
        }
        *(v4h*)(As + e0) = hv;
      }
      // ---- stage B: 4 tiles x 4 elems/thread ----
#pragma unroll
      for (int g = 0; g < 4; ++g) {
        int n = nblk + g * 16 + srow + r - pad;   // conv column shift
        bool nv = ((unsigned)n < (unsigned)N);
        int nc = nv ? n : 0;
        const float* bp = Bz + (long)nc * b_is;
        v4h hv;
        if (b_ks == 1) {                          // contiguous K: global_load_b128
          int kc = (kk0 <= K - 4) ? kk0 : ((K >= 4) ? K - 4 : 0);
          v4f f4 = *(const v4f*)(bp + kc);
#pragma unroll
          for (int i = 0; i < 4; ++i) {
            bool v = nv && (kk0 + i < K);
            hv[i] = (_Float16)(v ? f4[i] : 0.0f);
          }
        } else {
          float f[4];
#pragma unroll
          for (int i = 0; i < 4; ++i) {
            int kk = kk0 + i;
            int kc = (kk < K) ? kk : (K - 1);
            f[i] = bp[(long)kc * b_ks];
          }
#pragma unroll
          for (int i = 0; i < 4; ++i) {
            bool v = nv && (kk0 + i < K);
            hv[i] = (_Float16)(v ? f[i] : 0.0f);
          }
        }
        *(v4h*)(Bs + g * 512 + e0) = hv;
      }
      __syncthreads();
      v16h af = *(const v16h*)(As + lane * 16);            // 2x ds_load_b128
      v16h bf = *(const v16h*)(Bs + wave * 512 + lane * 16);
      acc = __builtin_amdgcn_wmma_f32_16x16x32_f16(false, af, false, bf,
                                                   (short)0, acc, false, false);
      __syncthreads();
    }
  }

  long cb = (long)z * c_bs;
  int n = nblk + wave * 16 + li;
  int mh = (lane >> 4) * 8;                       // C/D layout: VGPR rr -> M = mh + rr
#pragma unroll
  for (int rr = 0; rr < 8; ++rr) {
    int m = m0 + mh + rr;
    if (m < M) {
      float v = acc[rr];
      if (bias) v += bias[m];
      long o = cb + (long)m * N + n;
      if (Res) v += Res[o];                       // read-before-write: safe if Res==Cm
      Cm[o] = v * alpha;
    }
  }
}

// ---------------------------------------------------------------------------
// out[b, j] = b[j] + sum_s style[b,s] * w[j,s]      (j in [0, 2C))
// ---------------------------------------------------------------------------
__global__ void affine_kernel(const float* __restrict__ s, const float* __restrict__ w,
                              const float* __restrict__ b, float* __restrict__ out,
                              int Bn, int S, int O) {
  int idx = blockIdx.x * blockDim.x + threadIdx.x;
  if (idx >= Bn * O) return;
  int bi = idx / O, j = idx % O;
  const float* sp = s + (long)bi * S;
  const float* wp = w + (long)j * S;
  float acc = b[j];
  for (int q = 0; q < S; ++q) acc += sp[q] * wp[q];
  out[idx] = acc;
}

// ---------------------------------------------------------------------------
// AdaLayerNorm over channels: x element (b,i,c) at x + b*xb + i*xi + c*xc
// output [B,C,L]: out[b*C*L + c*L + i] = (1+gamma[c]) * ln + beta[c]
// ---------------------------------------------------------------------------
__global__ void adaln_kernel(const float* __restrict__ x, long xb, long xi, long xc,
                             const float* __restrict__ gbv, float* __restrict__ out,
                             int L, int C) {
  int row = blockIdx.x;
  int b = row / L, i = row % L;
  const float* xp = x + (long)b * xb + (long)i * xi;
  int tid = threadIdx.x;
  __shared__ float s1[256], s2[256];
  float a = 0.f, a2 = 0.f;
  for (int c = tid; c < C; c += blockDim.x) { float v = xp[(long)c * xc]; a += v; a2 += v * v; }
  s1[tid] = a; s2[tid] = a2; __syncthreads();
  for (int o = 128; o > 0; o >>= 1) {
    if (tid < o) { s1[tid] += s1[tid + o]; s2[tid] += s2[tid + o]; }
    __syncthreads();
  }
  float mu = s1[0] / C;
  float var = s2[0] / C - mu * mu;
  float rstd = rsqrtf(var + 1e-5f);
  const float* g  = gbv + (long)b * 2 * C;
  const float* be = g + C;
  float* op = out + (long)b * C * L + i;
  for (int c = tid; c < C; c += blockDim.x) {
    float v = (xp[(long)c * xc] - mu) * rstd;
    op[(long)c * L] = (1.f + g[c]) * v + be[c];
  }
}

// ---------------------------------------------------------------------------
// AdaIN1d over time (+ optional LeakyReLU 0.2). x,out: [B,C,L]. block per (b,c)
// ---------------------------------------------------------------------------
__global__ void adain_kernel(const float* __restrict__ x, const float* __restrict__ gbv,
                             float* __restrict__ out, int C, int L, int leak) {
  int bc = blockIdx.x;
  int b = bc / C, c = bc % C;
  const float* xp = x + (long)bc * L;
  float* op = out + (long)bc * L;
  int tid = threadIdx.x;
  __shared__ float s1[256], s2[256];
  float a = 0.f, a2 = 0.f;
  for (int i = tid; i < L; i += blockDim.x) { float v = xp[i]; a += v; a2 += v * v; }
  s1[tid] = a; s2[tid] = a2; __syncthreads();
  for (int o = 128; o > 0; o >>= 1) {
    if (tid < o) { s1[tid] += s1[tid + o]; s2[tid] += s2[tid + o]; }
    __syncthreads();
  }
  float mu = s1[0] / L;
  float var = s2[0] / L - mu * mu;
  float rstd = rsqrtf(var + 1e-5f);
  float g  = gbv[(long)b * 2 * C + c];
  float be = gbv[(long)b * 2 * C + C + c];
  for (int i = tid; i < L; i += blockDim.x) {
    float y = (1.f + g) * ((xp[i] - mu) * rstd) + be;
    if (leak) y = (y >= 0.f) ? y : 0.2f * y;
    op[i] = y;
  }
}

// ---------------------------------------------------------------------------
// tau[b,f] = argmax_t alignment[b,t,f]
// ---------------------------------------------------------------------------
__global__ void argmax_kernel(const float* __restrict__ al, int* __restrict__ tau,
                              int Bn, int T, int F) {
  int idx = blockIdx.x * blockDim.x + threadIdx.x;
  if (idx >= Bn * F) return;
  int b = idx / F, f = idx % F;
  const float* p = al + (long)b * T * F + f;
  float best = p[0];
  int bi = 0;
  for (int t = 1; t < T; ++t) {
    float v = p[(long)t * F];
    if (v > best) { best = v; bi = t; }
  }
  tau[idx] = bi;
}

// ---------------------------------------------------------------------------
// band-mask + softmax over T (in place). blockDim == T == 256. row = (b,h,f)
// ---------------------------------------------------------------------------
__global__ void softmax_kernel(float* __restrict__ sc, const int* __restrict__ tau,
                               const int* __restrict__ tlen, int Hh, int F, int T, int W) {
  int row = blockIdx.x;
  int f = row % F;
  int b = row / (Hh * F);
  float* p = sc + (long)row * T;
  int t = threadIdx.x;
  int tv = tau[(long)b * F + f];
  bool blocked = (t < tv - W) || (t > tv + W) || (t >= tlen[b]);
  float v = blocked ? -1e4f : p[t];
  __shared__ float sh[256];
  sh[t] = v; __syncthreads();
  for (int o = 128; o > 0; o >>= 1) {
    if (t < o) sh[t] = fmaxf(sh[t], sh[t + o]);
    __syncthreads();
  }
  float mx = sh[0]; __syncthreads();
  float e = expf(v - mx);
  sh[t] = e; __syncthreads();
  for (int o = 128; o > 0; o >>= 1) {
    if (t < o) sh[t] += sh[t + o];
    __syncthreads();
  }
  p[t] = e / sh[0];
}

// ---------------------------------------------------------------------------
// depthwise conv k=5 pad=2 + bias + SiLU. x,out: [B,C,F]
// ---------------------------------------------------------------------------
__global__ void dwsilu_kernel(const float* __restrict__ x, const float* __restrict__ w,
                              const float* __restrict__ b, float* __restrict__ out,
                              int C, int F, long total) {
  long idx = (long)blockIdx.x * blockDim.x + threadIdx.x;
  if (idx >= total) return;
  int f = (int)(idx % F);
  long bc = idx / F;
  int c = (int)(bc % C);
  const float* xp = x + (idx - f);
  float acc = b[c];
#pragma unroll
  for (int r = 0; r < 5; ++r) {
    int ff = f + r - 2;
    if ((unsigned)ff < (unsigned)F) acc += w[c * 5 + r] * xp[ff];
  }
  out[idx] = acc / (1.f + expf(-acc));  // silu
}

// ---------------------------------------------------------------------------
// out[b,f] = b0 + sum_c w[c] * x[b,c,f]
// ---------------------------------------------------------------------------
__global__ void proj_kernel(const float* __restrict__ x, const float* __restrict__ w,
                            const float* __restrict__ b, float* __restrict__ out,
                            int C, int F, int total) {
  int idx = blockIdx.x * blockDim.x + threadIdx.x;
  if (idx >= total) return;
  int bi = idx / F, f = idx % F;
  const float* xp = x + (long)bi * C * F + f;
  float acc = b[0];
  for (int c = 0; c < C; ++c) acc += w[c] * xp[(long)c * F];
  out[idx] = acc;
}

// ---------------------------------------------------------------------------
// host
// ---------------------------------------------------------------------------
extern "C" void kernel_launch(void* const* d_in, const int* in_sizes, int n_in,
                              void* d_out, int out_size, void* d_ws, size_t ws_size,
                              hipStream_t stream) {
  (void)in_sizes; (void)n_in; (void)out_size; (void)ws_size;
  const float* prosody = (const float*)d_in[0];   // [B,T,C]
  const float* align   = (const float*)d_in[1];   // [B,T,F]
  const float* style   = (const float*)d_in[2];   // [B,S]
  const int*   tlen    = (const int*)d_in[3];     // [B]
  // params flattened in dict-insertion order starting at d_in[4]
  auto prm = [&](int i) -> const float* { return (const float*)d_in[4 + i]; };
  // 0:qn_w 1:qn_b 2:kn_w 3:kn_b 4:q_w 5:q_b 6:k_w 7:k_b 8:v_w 9:v_b 10:o_w 11:o_b
  // 12:dw_w 13:dw_b 14:pw_w 15:pw_b
  // 16+8i: f0 block i (fc1_w,fc1_b,conv1_w,conv1_b,fc2_w,fc2_b,conv2_w,conv2_b)
  // 40+8i: n  block i (same)
  // 64:f0_proj_w 65:f0_proj_b 66:n_proj_w 67:n_proj_b

  float* ws = (float*)d_ws;
  const long GBn = (long)BB * 2 * CC;       // 9216 per affine
  const long BCF = (long)BB * CC * FF;      // 4,718,592
  const long BCT = (long)BB * CC * TT;      // 1,179,648
  const long SCn = (long)BB * HH * FF * TT; // 16,777,216
  float* gbuf = ws;                 // 14 style affines
  float* base = gbuf + 14 * GBn;    // base / x (reused in place)
  float* tA   = base + BCF;
  float* tB   = tA + BCF;
  float* kn   = tB + BCF;           // [B,C,T]
  float* k2   = kn + BCT;
  float* v2   = k2 + BCT;
  float* SCb  = v2 + BCT;           // scores [B,H,F,T]; later reused as p0/p1
  float* p0   = SCb;
  float* p1   = SCb + BCF;
  int*   tau  = (int*)(SCb + SCn);  // [B,F]

  // ---- style-conditioned affine params (14 of them) ----
  auto aff = [&](int slot, const float* w, const float* b) {
    affine_kernel<<<dim3((BB * 2 * CC + 255) / 256), 256, 0, stream>>>(
        style, w, b, gbuf + (long)slot * GBn, BB, SS, 2 * CC);
  };
  aff(0, prm(0), prm(1));  // qn
  aff(1, prm(2), prm(3));  // kn
  for (int i = 0; i < 3; ++i) {
    aff(2 + 2 * i, prm(16 + 8 * i + 0), prm(16 + 8 * i + 1));  // f0 fc1
    aff(3 + 2 * i, prm(16 + 8 * i + 4), prm(16 + 8 * i + 5));  // f0 fc2
    aff(8 + 2 * i, prm(40 + 8 * i + 0), prm(40 + 8 * i + 1));  // n  fc1
    aff(9 + 2 * i, prm(40 + 8 * i + 4), prm(40 + 8 * i + 5));  // n  fc2
  }

  // N must be a multiple of 64 at every call site (1024 or 256 here)
  auto gemm = [&](const float* A, long a_is, long a_ks, long a_bs, long a_rs,
                  const float* Bmat, long b_is, long b_ks, long b_bs,
                  const float* bias, const float* res, float* Cmat, long c_bs,
                  int M, int N, int K, int Z, int R, int pad, float alpha) {
    dim3 grid(N / 64, (M + 15) / 16, Z);
    wmma_gemm_kernel<<<grid, 128, 0, stream>>>(A, Bmat, bias, res, Cmat, M, N, K,
        a_is, a_ks, a_bs, a_rs, b_is, b_ks, b_bs, c_bs, R, pad, alpha);
  };

  const float inv_s2 = 0.70710678118654752f;

  // base[b,c,f] = sum_t prosody[b,t,c] * alignment[b,t,f]
  gemm(prosody, 1, CC, (long)TT * CC, 0,
       align,   1, FF, (long)TT * FF,
       nullptr, nullptr, base, (long)CC * FF,
       CC, FF, TT, BB, 1, 0, 1.0f);

  // query = AdaLN(base, over C per f) -> tA [B,C,F] ; key = AdaLN(prosody) -> kn [B,C,T]
  adaln_kernel<<<BB * FF, 256, 0, stream>>>(base, (long)CC * FF, 1L, (long)FF,
                                            gbuf + 0 * GBn, tA, FF, CC);
  adaln_kernel<<<BB * TT, 256, 0, stream>>>(prosody, (long)TT * CC, (long)CC, 1L,
                                            gbuf + 1 * GBn, kn, TT, CC);

  // q = Wq @ query + bq -> tB [B,C,F]
  gemm(prm(4), CC, 1, 0, 0, tA, 1, FF, (long)CC * FF,
       prm(5), nullptr, tB, (long)CC * FF, CC, FF, CC, BB, 1, 0, 1.0f);
  // k = Wk @ key -> k2 ; v = Wv @ key -> v2   [B,C,T]
  gemm(prm(6), CC, 1, 0, 0, kn, 1, TT, (long)CC * TT,
       prm(7), nullptr, k2, (long)CC * TT, CC, TT, CC, BB, 1, 0, 1.0f);
  gemm(prm(8), CC, 1, 0, 0, kn, 1, TT, (long)CC * TT,
       prm(9), nullptr, v2, (long)CC * TT, CC, TT, CC, BB, 1, 0, 1.0f);

  // band-mask argmax
  argmax_kernel<<<(BB * FF + 255) / 256, 256, 0, stream>>>(align, tau, BB, TT, FF);

  // scores[z=b*H+h, f, t] = (q^T k) / sqrt(d)
  gemm(tB, 1, FF, (long)DD * FF, 0,
       k2, 1, TT, (long)DD * TT,
       nullptr, nullptr, SCb, (long)FF * TT,
       FF, TT, DD, BB * HH, 1, 0, 1.0f / sqrtf((float)DD));

  // mask + softmax in place
  softmax_kernel<<<BB * HH * FF, 256, 0, stream>>>(SCb, tau, tlen, HH, FF, TT, WIN);

  // out[z, d, f] = sum_t v[z,d,t] * attn[z,f,t] -> tA [B,C,F]
  gemm(v2, TT, 1, (long)DD * TT, 0,
       SCb, TT, 1, (long)FF * TT,
       nullptr, nullptr, tA, (long)DD * FF,
       DD, FF, TT, BB * HH, 1, 0, 1.0f);

  // output projection -> tB
  gemm(prm(10), CC, 1, 0, 0, tA, 1, FF, (long)CC * FF,
       prm(11), nullptr, tB, (long)CC * FF, CC, FF, CC, BB, 1, 0, 1.0f);

  // depthwise k=5 + SiLU: tB -> tA
  dwsilu_kernel<<<(int)((BCF + 255) / 256), 256, 0, stream>>>(tB, prm(12), prm(13), tA,
                                                              CC, FF, BCF);

  // x = (base + pw(tA)) / sqrt(2), written in place over base
  gemm(prm(14), CC, 1, 0, 0, tA, 1, FF, (long)CC * FF,
       prm(15), base, base, (long)CC * FF, CC, FF, CC, BB, 1, 0, inv_s2);

  // ---- res-block branches ----
  auto branch = [&](int pb, int gs, const float* projw, const float* projb, float* outp) {
    const float* cur = base;  // x
    float* pouts[2] = {p0, p1};
    for (int i = 0; i < 3; ++i) {
      adain_kernel<<<BB * CC, 256, 0, stream>>>(cur, gbuf + (long)(gs + 2 * i) * GBn,
                                                tA, CC, FF, 1);
      gemm(prm(pb + 8 * i + 2), (long)CC * 3, 3, 0, 1,
           tA, 1, FF, (long)CC * FF,
           prm(pb + 8 * i + 3), nullptr, tB, (long)CC * FF,
           CC, FF, CC, BB, 3, 1, 1.0f);
      adain_kernel<<<BB * CC, 256, 0, stream>>>(tB, gbuf + (long)(gs + 2 * i + 1) * GBn,
                                                tA, CC, FF, 1);
      gemm(prm(pb + 8 * i + 6), (long)CC * 3, 3, 0, 1,
           tA, 1, FF, (long)CC * FF,
           prm(pb + 8 * i + 7), cur, pouts[i & 1], (long)CC * FF,
           CC, FF, CC, BB, 3, 1, inv_s2);
      cur = pouts[i & 1];
    }
    proj_kernel<<<(BB * FF + 255) / 256, 256, 0, stream>>>(cur, projw, projb, outp,
                                                           CC, FF, BB * FF);
  };

  float* out = (float*)d_out;
  branch(16, 2, prm(64), prm(65), out);            // f0 -> out[0 : B*F)
  branch(40, 8, prm(66), prm(67), out + BB * FF);  // n  -> out[B*F : 2*B*F)
}
